// Attention_51951924412498
// MI455X (gfx1250) — matrix operand order repaired and checked by reading, not compiled, not run on hardware
//
#include <hip/hip_runtime.h>
#include <stdint.h>

typedef __attribute__((ext_vector_type(16))) _Float16 v16h;
typedef __attribute__((ext_vector_type(8)))  _Float16 h8;
typedef __attribute__((ext_vector_type(8)))  float    v8f;
typedef int i4v  __attribute__((ext_vector_type(4)));
typedef int i8vi __attribute__((ext_vector_type(8)));

#define BATCH   4
#define SEQ     1024
#define DIM     1024
#define HEADS   16
#define DHEAD   64
#define INNER   1024
#define SCALE_Q 0.125f   // 64^-0.5

// ---------- CDNA5 async global->LDS copy (16B per lane, ASYNCcnt) ----------

__device__ __forceinline__ void async_copy_b128(void* lds_dst, const void* gsrc) {
  unsigned lds = (unsigned)(uintptr_t)lds_dst;           // generic->LDS offset (addr[31:0])
  unsigned long long ga = (unsigned long long)(uintptr_t)gsrc;
  asm volatile("global_load_async_to_lds_b128 %0, %1, off"
               :: "v"(lds), "v"(ga) : "memory");
}

__device__ __forceinline__ void wait_async0() {
  asm volatile("s_wait_asynccnt 0x0" ::: "memory");
}

// ---------- CDNA5 Tensor Data Mover: 2D tile global->LDS (TENSORcnt) ----------
// D# per CDNA5 ISA: group0 {count,lds_addr,global_addr,type=2},
// group1 {data_size=2B, tensor_dim0/1, tile_dim0/1, tensor_dim0_stride}. 2D => groups 2/3 NULL.
__device__ __forceinline__ void tdm_load_tile_2d(unsigned lds_addr, unsigned long long gaddr,
                                                 unsigned tensor_d0, unsigned tensor_d1,
                                                 unsigned tile_d0, unsigned tile_d1,
                                                 unsigned stride_d0) {
  i4v g0;
  g0[0] = 1;                                              // count=1, user mode
  g0[1] = (int)lds_addr;                                  // lds_addr (bytes)
  g0[2] = (int)(unsigned)gaddr;                           // global_addr[31:0]
  g0[3] = (int)((unsigned)(gaddr >> 32) | (2u << 30));    // global_addr[56:32] | type=2
  i8vi g1;
  g1[0] = (int)(1u << 16);                                // data_size=1 (2 bytes), mask=0
  g1[1] = (int)((tensor_d0 & 0xFFFFu) << 16);             // tensor_dim0[15:0]
  g1[2] = (int)((tensor_d0 >> 16) | ((tensor_d1 & 0xFFFFu) << 16)); // d0[31:16]|d1[15:0]
  g1[3] = (int)((tensor_d1 >> 16) | (tile_d0 << 16));     // d1[31:16]|tile_dim0
  g1[4] = (int)tile_d1;                                   // tile_dim1, tile_dim2=0
  g1[5] = (int)stride_d0;                                 // tensor_dim0_stride[31:0]
  g1[6] = 0;                                              // stride hi, dim1_stride lo
  g1[7] = 0;
  asm volatile("tensor_load_to_lds %0, %1" :: "s"(g0), "s"(g1) : "memory");
}

// ---------- fragment loaders (layouts per CDNA5 ISA) ----------

// A-matrix 16x32 f16: lanes 0-15 row m=lane, K 0..7 | 16..23; lanes 16-31: K 8..15 | 24..31
__device__ __forceinline__ v16h fragA(const _Float16* p, int hi) {
  h8 lo = *(const h8*)(p + hi * 8);
  h8 hv = *(const h8*)(p + 16 + hi * 8);
  v16h r;
#pragma unroll
  for (int i = 0; i < 8; ++i) { r[i] = lo[i]; r[i + 8] = hv[i]; }
  return r;
}

// B-matrix 32x16 f16: lane&15 = column n; lanes 0-15: K 0..7 | 16..23; lanes 16-31: K 8..15 | 24..31
__device__ __forceinline__ v16h fragB(const _Float16* p, int hi) {
  h8 lo = *(const h8*)(p + hi * 8);
  h8 hv = *(const h8*)(p + 16 + hi * 8);
  v16h r;
#pragma unroll
  for (int i = 0; i < 8; ++i) { r[i] = lo[i]; r[i + 8] = hv[i]; }
  return r;
}

// ---------- conversion kernels ----------

__global__ void cvt_f32_f16(const float* __restrict__ s, _Float16* __restrict__ d, int n) {
  int i = blockIdx.x * blockDim.x + threadIdx.x;
  if (i < n) d[i] = (_Float16)s[i];
}

__global__ void cvt_mask_u8(const int* __restrict__ s, unsigned char* __restrict__ d, int n) {
  int i = blockIdx.x * blockDim.x + threadIdx.x;
  if (i < n) d[i] = (s[i] != 0) ? 1 : 0;
}

// ---------- tiled WMMA GEMM: C[MxN] = alpha * A[MxK] @ B[KxN] (+bias) ----------
// N, K compile-time; F16OUT selects f16 store vs f32+bias store (no runtime branching).
// A tile staged by the Tensor Data Mover (one descriptor per k-step, issued by wave 0);
// B tile staged transposed through VGPRs.
template<int N, int K, bool F16OUT>
__global__ __launch_bounds__(128) void gemm_wmma(
    const _Float16* __restrict__ A, const _Float16* __restrict__ B,
    int M, float alpha,
    _Float16* __restrict__ Ch, float* __restrict__ Cf, const float* __restrict__ bias)
{
  __shared__ _Float16 As[64][32];   // [m][k]  (TDM-staged, 4KB linear)
  __shared__ _Float16 Bt[64][32];   // [n][k]  (transposed stage for contiguous B frags)

  const int tid    = threadIdx.x;
  const int lane   = tid & 31;
  const int laneLo = lane & 15;
  const int laneHi = lane >> 4;
  const int wave   = tid >> 5;
  const int wy = (wave >> 1) * 32;
  const int wx = (wave & 1) * 32;
  const int mBase = blockIdx.y * 64;
  const int nBase = blockIdx.x * 64;

  const unsigned ldsAs = (unsigned)(uintptr_t)&As[0][0];
  unsigned long long aTile = (unsigned long long)(uintptr_t)A + ((size_t)mBase * K) * 2;

  // B staging: chunk c -> k-row c>>3, n-offset (c&7)*8
  const int kB0 = tid >> 3,         nB0 = (tid & 7) << 3;
  const int kB1 = (tid + 128) >> 3, nB1 = ((tid + 128) & 7) << 3;
  const _Float16* bSrc0 = B + (size_t)kB0 * N + nBase + nB0;
  const _Float16* bSrc1 = B + (size_t)kB1 * N + nBase + nB1;
  const size_t bStep = (size_t)32 * N;

  v8f acc[2][2] = {};

  for (int kb = 0; kb < K; kb += 32) {
    // TDM: pull the 64x32 f16 A tile (row stride K) straight into LDS
    if (wave == 0)
      tdm_load_tile_2d(ldsAs, aTile, K, (unsigned)M, 32, 64, K);
    __builtin_prefetch(bSrc0 + bStep);
    // B tile transposed through VGPRs
    h8 b0 = *(const h8*)bSrc0;
    h8 b1 = *(const h8*)bSrc1;
#pragma unroll
    for (int j = 0; j < 8; ++j) { Bt[nB0 + j][kB0] = b0[j]; Bt[nB1 + j][kB1] = b1[j]; }
    __builtin_amdgcn_s_wait_tensorcnt(0);
    __syncthreads();

    v16h aF[2], bF[2];
#pragma unroll
    for (int i = 0; i < 2; ++i) aF[i] = fragA(&As[wy + i * 16 + laneLo][0], laneHi);
#pragma unroll
    for (int j = 0; j < 2; ++j) bF[j] = fragB(&Bt[wx + j * 16 + laneLo][0], laneHi);
#pragma unroll
    for (int i = 0; i < 2; ++i)
#pragma unroll
      for (int j = 0; j < 2; ++j)
        acc[i][j] = __builtin_amdgcn_wmma_f32_16x16x32_f16(
            false, aF[i], false, bF[j], (short)0, acc[i][j], false, false);
    __syncthreads();

    aTile += 64;                       // 32 f16 = 64 bytes along K
    bSrc0 += bStep; bSrc1 += bStep;
  }

  // uniform epilogue (compile-time specialized, immediate-offset stores)
  if constexpr (F16OUT) {
    _Float16* outp = Ch + (size_t)(mBase + wy + laneHi * 8) * N + nBase + wx + laneLo;
#pragma unroll
    for (int i = 0; i < 2; ++i)
#pragma unroll
      for (int j = 0; j < 2; ++j)
#pragma unroll
        for (int v = 0; v < 8; ++v)
          outp[(size_t)(i * 16 + v) * N + j * 16] = (_Float16)(acc[i][j][v] * alpha);
  } else {
    float* outp = Cf + (size_t)(mBase + wy + laneHi * 8) * N + nBase + wx + laneLo;
    float bv[2];
#pragma unroll
    for (int j = 0; j < 2; ++j) bv[j] = bias[nBase + wx + j * 16 + laneLo];
#pragma unroll
    for (int i = 0; i < 2; ++i)
#pragma unroll
      for (int j = 0; j < 2; ++j)
#pragma unroll
        for (int v = 0; v < 8; ++v)
          outp[(size_t)(i * 16 + v) * N + j * 16] = acc[i][j][v] * alpha + bv[j];
  }
}

// ---------- flash-attention kernel ----------
// grid = (SEQ/64, HEADS, BATCH); 128 threads = 4 waves, each wave owns 16 query rows.
// Q already carries SCALE_Q. Q/K/V are row-major [B*SEQ][INNER]; head h = cols h*64..h*64+63.
__global__ __launch_bounds__(128) void attn_wmma(
    const _Float16* __restrict__ Q, const _Float16* __restrict__ Km,
    const _Float16* __restrict__ Vm, const unsigned char* __restrict__ maskb,
    _Float16* __restrict__ Oh)
{
  __shared__ _Float16 Ks[32][64];       // [key][d] (async-staged)
  __shared__ _Float16 Vt[64][32];       // [d][key] (transposed through VGPRs)
  __shared__ _Float16 Ps[4][16][32];    // per-wave P scratch [m][k]

  const int tid    = threadIdx.x;
  const int lane   = tid & 31;
  const int laneLo = lane & 15;
  const int laneHi = lane >> 4;
  const int wave   = tid >> 5;
  const int h  = blockIdx.y;
  const int b  = blockIdx.z;
  const int qRow0 = blockIdx.x * 64 + wave * 16;

  // staging assignments: 256 chunks of 8 halves; chunk c -> key c>>3, d-offset (c&7)*8
  const int key0 = tid >> 3,         d80 = (tid & 7) << 3;
  const int key1 = (tid + 128) >> 3, d81 = ((tid + 128) & 7) << 3;
  const _Float16* kSrc0 = Km + (size_t)(b * SEQ + key0) * INNER + h * DHEAD + d80;
  const _Float16* kSrc1 = Km + (size_t)(b * SEQ + key1) * INNER + h * DHEAD + d81;
  const _Float16* vSrc0 = Vm + (size_t)(b * SEQ + key0) * INNER + h * DHEAD + d80;
  const _Float16* vSrc1 = Vm + (size_t)(b * SEQ + key1) * INNER + h * DHEAD + d81;
  _Float16* kDst0 = &Ks[key0][d80];
  _Float16* kDst1 = &Ks[key1][d81];
  const size_t jStep = (size_t)32 * INNER;

  // preload Q fragments (16 rows x 64 d = 2 A-frags), scale already folded in
  v16h qa[2];
#pragma unroll
  for (int f = 0; f < 2; ++f)
    qa[f] = fragA(Q + (size_t)(b * SEQ + qRow0 + laneLo) * INNER + h * DHEAD + f * 32, laneHi);

  float rmax[8], rsum[8];
#pragma unroll
  for (int v = 0; v < 8; ++v) { rmax[v] = -3.0e38f; rsum[v] = 0.0f; }
  v8f o[4] = {};

  for (int jBase = 0; jBase < SEQ; jBase += 32) {
    // async K tile copy; V transposed via VGPRs; prefetch next tiles
    async_copy_b128(kDst0, kSrc0);
    async_copy_b128(kDst1, kSrc1);
    __builtin_prefetch(kSrc0 + jStep);
    __builtin_prefetch(vSrc0 + jStep);
    h8 v0 = *(const h8*)vSrc0;
    h8 v1 = *(const h8*)vSrc1;
#pragma unroll
    for (int j = 0; j < 8; ++j) { Vt[d80 + j][key0] = v0[j]; Vt[d81 + j][key1] = v1[j]; }
    wait_async0();
    __syncthreads();

    // S = Q K^T for 16x32 score strip (two 16x16 tiles, K-dim 64 = 2 steps)
    v8f st[2];
#pragma unroll
    for (int nt = 0; nt < 2; ++nt) {
      v8f s = {};
#pragma unroll
      for (int ks = 0; ks < 2; ++ks) {
        v16h bK = fragB(&Ks[nt * 16 + laneLo][ks * 32], laneHi);
        s = __builtin_amdgcn_wmma_f32_16x16x32_f16(false, qa[ks], false, bK,
                                                   (short)0, s, false, false);
      }
      st[nt] = s;
    }

    // mask + online softmax (row stats per the 8 C-rows this lane covers)
#pragma unroll
    for (int v = 0; v < 8; ++v) {
      int qr = qRow0 + laneHi * 8 + v;
      const unsigned char* mrow = maskb + (size_t)(b * SEQ + qr) * SEQ + jBase;
#pragma unroll
      for (int nt = 0; nt < 2; ++nt)
        if (mrow[nt * 16 + laneLo]) st[nt][v] = -3.0e38f;

      float tm = fmaxf(st[0][v], st[1][v]);
      tm = fmaxf(tm, __shfl_xor(tm, 1));
      tm = fmaxf(tm, __shfl_xor(tm, 2));
      tm = fmaxf(tm, __shfl_xor(tm, 4));
      tm = fmaxf(tm, __shfl_xor(tm, 8));

      float nm   = fmaxf(rmax[v], tm);
      float corr = __expf(rmax[v] - nm);
      rmax[v] = nm;

      float ls = 0.0f;
#pragma unroll
      for (int nt = 0; nt < 2; ++nt) {
        float p = __expf(st[nt][v] - nm);
        st[nt][v] = p;
        ls += p;
      }
      ls += __shfl_xor(ls, 1);
      ls += __shfl_xor(ls, 2);
      ls += __shfl_xor(ls, 4);
      ls += __shfl_xor(ls, 8);
      rsum[v] = rsum[v] * corr + ls;
#pragma unroll
      for (int t = 0; t < 4; ++t) o[t][v] *= corr;
    }

    // write P (C-layout) to LDS, re-read in A-layout for P·V
#pragma unroll
    for (int v = 0; v < 8; ++v)
#pragma unroll
      for (int nt = 0; nt < 2; ++nt)
        Ps[wave][laneHi * 8 + v][nt * 16 + laneLo] = (_Float16)st[nt][v];
    __syncthreads();

    v16h pa = fragA(&Ps[wave][laneLo][0], laneHi);
#pragma unroll
    for (int t = 0; t < 4; ++t) {
      v16h bV = fragB(&Vt[t * 16 + laneLo][0], laneHi);
      o[t] = __builtin_amdgcn_wmma_f32_16x16x32_f16(false, pa, false, bV,
                                                    (short)0, o[t], false, false);
    }
    __syncthreads();

    kSrc0 += jStep; kSrc1 += jStep;
    vSrc0 += jStep; vSrc1 += jStep;
  }

  // normalize and store f16 attention output, row-major [B*SEQ][INNER]
  _Float16* outp = Oh + (size_t)(b * SEQ + qRow0 + laneHi * 8) * INNER + h * DHEAD + laneLo;
#pragma unroll
  for (int v = 0; v < 8; ++v) {
    float inv = 1.0f / (rsum[v] + 1e-20f);
#pragma unroll
    for (int t = 0; t < 4; ++t)
      outp[(size_t)v * INNER + t * 16] = (_Float16)(o[t][v] * inv);
  }
}

// ---------- host launch ----------

extern "C" void kernel_launch(void* const* d_in, const int* in_sizes, int n_in,
                              void* d_out, int out_size, void* d_ws, size_t ws_size,
                              hipStream_t stream) {
  const float* x    = (const float*)d_in[0];
  const int*   mask = (const int*)d_in[1];
  const float* Wq   = (const float*)d_in[2];
  const float* Wk   = (const float*)d_in[3];
  const float* Wv   = (const float*)d_in[4];
  const float* Wo   = (const float*)d_in[5];
  const float* bo   = (const float*)d_in[6];
  float* out = (float*)d_out;

  char* ws = (char*)d_ws;
  size_t off = 0;
  auto alloc = [&](size_t bytes) {
    void* p = ws + off;
    off += (bytes + 255) & ~(size_t)255;
    return p;
  };

  const int BN = BATCH * SEQ;                       // 4096
  _Float16* xh  = (_Float16*)alloc((size_t)BN * DIM * 2);
  _Float16* wqh = (_Float16*)alloc((size_t)DIM * INNER * 2);
  _Float16* wkh = (_Float16*)alloc((size_t)DIM * INNER * 2);
  _Float16* wvh = (_Float16*)alloc((size_t)DIM * INNER * 2);
  _Float16* woh = (_Float16*)alloc((size_t)INNER * DIM * 2);
  _Float16* qh  = (_Float16*)alloc((size_t)BN * INNER * 2);
  _Float16* kh  = (_Float16*)alloc((size_t)BN * INNER * 2);
  _Float16* vh  = (_Float16*)alloc((size_t)BN * INNER * 2);
  _Float16* aoh = (_Float16*)alloc((size_t)BN * INNER * 2);
  unsigned char* mb = (unsigned char*)alloc((size_t)BATCH * SEQ * SEQ);

  const int nX = BN * DIM, nW = DIM * INNER, nM = BATCH * SEQ * SEQ;
  cvt_f32_f16<<<(nX + 255) / 256, 256, 0, stream>>>(x,  xh,  nX);
  cvt_f32_f16<<<(nW + 255) / 256, 256, 0, stream>>>(Wq, wqh, nW);
  cvt_f32_f16<<<(nW + 255) / 256, 256, 0, stream>>>(Wk, wkh, nW);
  cvt_f32_f16<<<(nW + 255) / 256, 256, 0, stream>>>(Wv, wvh, nW);
  cvt_f32_f16<<<(nW + 255) / 256, 256, 0, stream>>>(Wo, woh, nW);
  cvt_mask_u8<<<(nM + 255) / 256, 256, 0, stream>>>(mask, mb, nM);

  dim3 gP(INNER / 64, BN / 64);   // (16, 64)
  gemm_wmma<INNER, DIM, true><<<gP, 128, 0, stream>>>(xh, wqh, BN, SCALE_Q, qh, nullptr, nullptr);
  gemm_wmma<INNER, DIM, true><<<gP, 128, 0, stream>>>(xh, wkh, BN, 1.0f,    kh, nullptr, nullptr);
  gemm_wmma<INNER, DIM, true><<<gP, 128, 0, stream>>>(xh, wvh, BN, 1.0f,    vh, nullptr, nullptr);

  dim3 gA(SEQ / 64, HEADS, BATCH); // (16, 16, 4)
  attn_wmma<<<gA, 128, 0, stream>>>(qh, kh, vh, mb, aoh);

  dim3 gO(DIM / 64, BN / 64);
  gemm_wmma<DIM, INNER, false><<<gO, 128, 0, stream>>>(aoh, woh, BN, 1.0f, nullptr, out, bo);
}
